// HKANGNN_9259949490519
// MI455X (gfx1250) — compile-verified, matrix operand order checked by reading, not run on hardware
//
#include <hip/hip_runtime.h>
#include <hip/hip_bf16.h>
#include <math.h>

// ---------- problem constants ----------
constexpr int NE = 100000;
constexpr int NU = 50000;
constexpr int NS = 20000;
constexpr int H  = 128;
constexpr int KE = 768;

// ---------- workspace layout (float elements) ----------
constexpr size_t OFF_HE    = 0;
constexpr size_t OFF_S1    = OFF_HE + (size_t)NE * H;
constexpr size_t OFF_C1    = OFF_S1 + (size_t)NE * H;
constexpr size_t OFF_S2    = OFF_C1 + (size_t)NE;
constexpr size_t OFF_C2    = OFF_S2 + (size_t)NE * H;
constexpr size_t OFF_HU    = OFF_C2 + (size_t)NE;
constexpr size_t OFF_HS    = OFF_HU + (size_t)NU * H;
constexpr size_t OFF_EMAIL = OFF_HS + (size_t)NS * H;
constexpr size_t OFF_BSUM  = OFF_EMAIL + (size_t)NE * H;          // 128 f32
constexpr size_t OFF_WPE   = OFF_BSUM + 128;                      // Wp_email bf16: 768*128/2 f32 slots
constexpr size_t OFF_WL1   = OFF_WPE + (size_t)KE * H / 2;        // Wl_ue bf16
constexpr size_t OFF_WL2   = OFF_WL1 + (size_t)H * H / 2;         // Wl_se bf16
constexpr size_t OFF_WS16  = OFF_WL2 + (size_t)H * H / 2;         // Wsum  bf16
constexpr size_t ZERO_N4   = ((size_t)2 * (NE * H + NE)) / 4;     // s1,c1,s2,c2 contiguous

// ---------- vector types (trivially-constructible, union-safe) ----------
typedef __attribute__((ext_vector_type(16))) __bf16       v16bf;
typedef __attribute__((ext_vector_type(8)))  float        v8f;
typedef __attribute__((ext_vector_type(4)))  float        f32x4;
typedef __attribute__((ext_vector_type(4)))  unsigned int u32x4;
typedef __attribute__((ext_vector_type(2)))  unsigned int u32x2;

union ABf { v16bf v; u32x4 q[2]; };

// Hardware f32->bf16 (RNE) via fptrunc; packs two values into one dword.
__device__ __forceinline__ unsigned int pack_bf16(float a, float b) {
  union { __bf16 h; unsigned short s; } ua, ub;
  ua.h = (__bf16)a;
  ub.h = (__bf16)b;
  return (unsigned int)ua.s | ((unsigned int)ub.s << 16);
}

// ---------- zero scratch (s1,c1,s2,c2) ----------
__global__ void k_zero(float* __restrict__ p, long n4) {
  long tid = (long)blockIdx.x * blockDim.x + threadIdx.x;
  if (tid < n4) {
    f32x4 z = {0.f, 0.f, 0.f, 0.f};
    *(f32x4*)(p + tid * 4) = z;
  }
}

// ---------- f32 -> packed bf16 weight conversion (4 dwords / thread) ----------
__global__ void k_cvt_bf16(const float* __restrict__ src, unsigned int* __restrict__ dst,
                           int n_qword) {   // n_qword = n_elems/8
  int tid = blockIdx.x * blockDim.x + threadIdx.x;
  if (tid >= n_qword) return;
  f32x4 f0 = *(const f32x4*)(src + (size_t)tid * 8);
  f32x4 f1 = *(const f32x4*)(src + (size_t)tid * 8 + 4);
  u32x4 q = {pack_bf16(f0.x, f0.y), pack_bf16(f0.z, f0.w),
             pack_bf16(f1.x, f1.y), pack_bf16(f1.z, f1.w)};
  *(u32x4*)(dst + (size_t)tid * 4) = q;
}

// ---------- Wsum = bf16(Wr_ue + Wr_se) ; bsum = bl_ue + bl_se (f32) ----------
__global__ void k_wsum(const float* __restrict__ Wr_ue, const float* __restrict__ Wr_se,
                       const float* __restrict__ bl_ue, const float* __restrict__ bl_se,
                       unsigned int* __restrict__ Wsum16, float* __restrict__ bsum) {
  int tid = blockIdx.x * blockDim.x + threadIdx.x;   // one dword pair per thread
  if (tid < H * H / 2) {
    float a = Wr_ue[2 * tid + 0] + Wr_se[2 * tid + 0];
    float b = Wr_ue[2 * tid + 1] + Wr_se[2 * tid + 1];
    Wsum16[tid] = pack_bf16(a, b);
  }
  if (tid < H) bsum[tid] = bl_ue[tid] + bl_se[tid];
}

// ---------- tiny projections (K=8 and K=1) ----------
__global__ void k_proj_url(const float* __restrict__ x, const float* __restrict__ W,
                           const float* __restrict__ b, float* __restrict__ hu) {
  int tid = blockIdx.x * blockDim.x + threadIdx.x;
  if (tid >= NU * H) return;
  int n = tid >> 7, h = tid & 127;
  const float* xr = x + (size_t)n * 8;
  const float* wr = W + (size_t)h * 8;
  float s = b[h];
#pragma unroll
  for (int k = 0; k < 8; ++k) s += xr[k] * wr[k];
  hu[tid] = s;
}

__global__ void k_proj_sender(const float* __restrict__ x, const float* __restrict__ W,
                              const float* __restrict__ b, float* __restrict__ hs) {
  int tid = blockIdx.x * blockDim.x + threadIdx.x;
  if (tid >= NS * H) return;
  int n = tid >> 7, h = tid & 127;
  hs[tid] = x[n] * W[h] + b[h];
}

// ---------- he = x_email @ Wp_email^T + bp  (WMMA bf16, K=768) ----------
// Block = 8 waves covering 16 rows x 128 cols; wave w owns 16-col tile w.
// A (16x768) staged once in LDS as bf16 (padded stride 1552B); B loaded as raw
// bf16 b128s from the pre-converted weight buffer.
__global__ void k_he_gemm(const float* __restrict__ X, const unsigned short* __restrict__ Wbf,
                          const float* __restrict__ bias, float* __restrict__ he) {
  __shared__ __align__(16) unsigned char lds[16 * 1552];
  const int rowbase = blockIdx.x * 16;
  const int t = threadIdx.x;
  {
    const int row = t >> 4;
    const int c0  = (t & 15) * 48;
    const float* src = X + (size_t)(rowbase + row) * KE + c0;
    unsigned char* dst = lds + row * 1552 + c0 * 2;
#pragma unroll
    for (int j = 0; j < 12; ++j) {
      f32x4 f = *(const f32x4*)(src + j * 4);
      u32x2 p = {pack_bf16(f.x, f.y), pack_bf16(f.z, f.w)};
      *(u32x2*)(dst + j * 8) = p;
    }
  }
  __syncthreads();

  const int lane = t & 31;
  const int wv   = t >> 5;
  const int nb   = wv * 16;
  const int r    = lane & 15;
  const int hi   = lane >> 4;

  float bias_v = bias[nb + r];
  v8f acc;
#pragma unroll
  for (int i = 0; i < 8; ++i) acc[i] = bias_v;

  const unsigned char* arow = lds + r * 1552;
  const unsigned short* wrow = Wbf + (size_t)(nb + r) * KE;

  for (int k = 0; k < KE; k += 32) {
    ABf a, b;
    a.q[0] = *(const u32x4*)(arow + (k + hi * 8) * 2);
    a.q[1] = *(const u32x4*)(arow + (k + 16 + hi * 8) * 2);
    b.q[0] = *(const u32x4*)(wrow + k + hi * 8);
    b.q[1] = *(const u32x4*)(wrow + k + 16 + hi * 8);
    acc = __builtin_amdgcn_wmma_f32_16x16x32_bf16(false, a.v, false, b.v,
                                                  (short)0, acc, false, false);
  }
#pragma unroll
  for (int v = 0; v < 8; ++v)
    he[(size_t)(rowbase + v + 8 * hi) * H + nb + r] = acc[v];
}

// ---------- edge scatter: one wave per edge, 4 float atomics per lane ----------
__global__ void k_scatter(const float* __restrict__ feat, const int* __restrict__ src,
                          const int* __restrict__ dst, int E,
                          float* __restrict__ s, float* __restrict__ c) {
  int gid = blockIdx.x * blockDim.x + threadIdx.x;
  int e = gid >> 5;
  if (e >= E) return;
  int lane = threadIdx.x & 31;
  int si = src[e], di = dst[e];
  f32x4 v = *(const f32x4*)(feat + (size_t)si * H + lane * 4);
  float* o = s + (size_t)di * H + lane * 4;
  atomicAdd(o + 0, v.x);
  atomicAdd(o + 1, v.y);
  atomicAdd(o + 2, v.z);
  atomicAdd(o + 3, v.w);
  if (lane == 0) atomicAdd(c + di, 1.0f);
}

// ---------- fused SAGE combine (3 accumulated WMMA GEMMs, K=128) ----------
// email = leaky_relu(0.5*(agg1@Wl_ue^T + agg2@Wl_se^T + he@Wsum^T + bsum), 0.2) + he
__global__ void k_combine(const float* __restrict__ s1, const float* __restrict__ c1,
                          const float* __restrict__ s2, const float* __restrict__ c2,
                          const float* __restrict__ he,
                          const unsigned short* __restrict__ Wl1, const unsigned short* __restrict__ Wl2,
                          const unsigned short* __restrict__ Ws, const float* __restrict__ bsum,
                          float* __restrict__ email) {
  __shared__ __align__(16) unsigned char lds[3 * 16 * 272];
  const int rowbase = blockIdx.x * 16;
  const int t = threadIdx.x;
  {
    const int row = t >> 4;
    const int c0  = (t & 15) * 8;
    float inv1 = 1.0f / fmaxf(c1[rowbase + row], 1.0f);
    float inv2 = 1.0f / fmaxf(c2[rowbase + row], 1.0f);
    const float* p1 = s1 + (size_t)(rowbase + row) * H + c0;
    const float* p2 = s2 + (size_t)(rowbase + row) * H + c0;
    const float* p3 = he + (size_t)(rowbase + row) * H + c0;
    unsigned char* d1 = lds + row * 272 + c0 * 2;
    unsigned char* d2 = d1 + 16 * 272;
    unsigned char* d3 = d2 + 16 * 272;
#pragma unroll
    for (int j = 0; j < 2; ++j) {
      f32x4 f1 = *(const f32x4*)(p1 + j * 4);
      f32x4 f2 = *(const f32x4*)(p2 + j * 4);
      f32x4 f3 = *(const f32x4*)(p3 + j * 4);
      u32x2 a = {pack_bf16(f1.x * inv1, f1.y * inv1), pack_bf16(f1.z * inv1, f1.w * inv1)};
      u32x2 b = {pack_bf16(f2.x * inv2, f2.y * inv2), pack_bf16(f2.z * inv2, f2.w * inv2)};
      u32x2 cc = {pack_bf16(f3.x, f3.y), pack_bf16(f3.z, f3.w)};
      *(u32x2*)(d1 + j * 8) = a;
      *(u32x2*)(d2 + j * 8) = b;
      *(u32x2*)(d3 + j * 8) = cc;
    }
  }
  __syncthreads();

  const int lane = t & 31;
  const int wv   = t >> 5;
  const int nb   = wv * 16;
  const int r    = lane & 15;
  const int hi   = lane >> 4;

  float bias_v = bsum[nb + r];
  v8f acc;
#pragma unroll
  for (int i = 0; i < 8; ++i) acc[i] = bias_v;

  const unsigned char* a1row = lds + r * 272;
  const unsigned char* a2row = a1row + 16 * 272;
  const unsigned char* a3row = a2row + 16 * 272;
  const unsigned short* w1row = Wl1 + (size_t)(nb + r) * H;
  const unsigned short* w2row = Wl2 + (size_t)(nb + r) * H;
  const unsigned short* w3row = Ws  + (size_t)(nb + r) * H;

#pragma unroll
  for (int k = 0; k < H; k += 32) {
    ABf a, b;
    a.q[0] = *(const u32x4*)(a1row + (k + hi * 8) * 2);
    a.q[1] = *(const u32x4*)(a1row + (k + 16 + hi * 8) * 2);
    b.q[0] = *(const u32x4*)(w1row + k + hi * 8);
    b.q[1] = *(const u32x4*)(w1row + k + 16 + hi * 8);
    acc = __builtin_amdgcn_wmma_f32_16x16x32_bf16(false, a.v, false, b.v,
                                                  (short)0, acc, false, false);
    a.q[0] = *(const u32x4*)(a2row + (k + hi * 8) * 2);
    a.q[1] = *(const u32x4*)(a2row + (k + 16 + hi * 8) * 2);
    b.q[0] = *(const u32x4*)(w2row + k + hi * 8);
    b.q[1] = *(const u32x4*)(w2row + k + 16 + hi * 8);
    acc = __builtin_amdgcn_wmma_f32_16x16x32_bf16(false, a.v, false, b.v,
                                                  (short)0, acc, false, false);
    a.q[0] = *(const u32x4*)(a3row + (k + hi * 8) * 2);
    a.q[1] = *(const u32x4*)(a3row + (k + 16 + hi * 8) * 2);
    b.q[0] = *(const u32x4*)(w3row + k + hi * 8);
    b.q[1] = *(const u32x4*)(w3row + k + 16 + hi * 8);
    acc = __builtin_amdgcn_wmma_f32_16x16x32_bf16(false, a.v, false, b.v,
                                                  (short)0, acc, false, false);
  }
#pragma unroll
  for (int v = 0; v < 8; ++v) {
    size_t m = (size_t)(rowbase + v + 8 * hi);
    float o = acc[v] * 0.5f;
    float lr = o > 0.0f ? o : 0.2f * o;
    float res = he[m * H + nb + r];
    email[m * H + nb + r] = lr + res;
  }
}

// ---------- KAN head: one wave per row, lane handles 4 features ----------
__global__ void k_kan(const float* __restrict__ email, const float* __restrict__ base_w,
                      const float* __restrict__ spline_w, float* __restrict__ out) {
  int gid = blockIdx.x * blockDim.x + threadIdx.x;
  int row = gid >> 5;
  if (row >= NE) return;
  int lane = threadIdx.x & 31;
  f32x4 xv = *(const f32x4*)(email + (size_t)row * H + lane * 4);
  float acc0 = 0.0f, acc1 = 0.0f;
#pragma unroll
  for (int j = 0; j < 4; ++j) {
    int i = lane * 4 + j;
    float x = xv[j];
    float sil = x / (1.0f + __expf(-x));               // silu
    acc0 += sil * base_w[i];
    acc1 += sil * base_w[H + i];
    // Cox-de Boor on uniform grid g(t) = 0.4*t - 2.2 (GS=5, K=3)
    float b[11];
#pragma unroll
    for (int tt = 0; tt < 11; ++tt) {
      float g0 = 0.4f * tt - 2.2f;
      float g1 = 0.4f * (tt + 1) - 2.2f;
      b[tt] = (x >= g0 && x < g1) ? 1.0f : 0.0f;
    }
#pragma unroll
    for (int p = 1; p <= 3; ++p) {
      float inv = 1.0f / (0.4f * p);
#pragma unroll
      for (int tt = 0; tt + p < 11; ++tt) {
        float gt   = 0.4f * tt - 2.2f;
        float gtp1 = 0.4f * (tt + p + 1) - 2.2f;
        b[tt] = (x - gt) * inv * b[tt] + (gtp1 - x) * inv * b[tt + 1];
      }
    }
    const float* sw0 = spline_w + (size_t)i * 8;
    const float* sw1 = spline_w + (size_t)H * 8 + (size_t)i * 8;
#pragma unroll
    for (int tt = 0; tt < 8; ++tt) {
      acc0 += b[tt] * sw0[tt];
      acc1 += b[tt] * sw1[tt];
    }
  }
#pragma unroll
  for (int off = 16; off > 0; off >>= 1) {
    acc0 += __shfl_down(acc0, off, 32);
    acc1 += __shfl_down(acc1, off, 32);
  }
  if (lane == 0) {
    out[(size_t)row * 2 + 0] = acc0;
    out[(size_t)row * 2 + 1] = acc1;
  }
}

extern "C" void kernel_launch(void* const* d_in, const int* in_sizes, int n_in,
                              void* d_out, int out_size, void* d_ws, size_t ws_size,
                              hipStream_t stream) {
  const float* x_email      = (const float*)d_in[0];
  const float* x_url        = (const float*)d_in[1];
  const float* x_sender     = (const float*)d_in[2];
  const int*   ei_rev_cont  = (const int*)d_in[4];   // url -> email
  const int*   ei_sends     = (const int*)d_in[5];   // sender -> email
  const float* Wp_email     = (const float*)d_in[7];
  const float* bp_email     = (const float*)d_in[8];
  const float* Wp_url       = (const float*)d_in[9];
  const float* bp_url       = (const float*)d_in[10];
  const float* Wp_sender    = (const float*)d_in[11];
  const float* bp_sender    = (const float*)d_in[12];
  const float* kan_base_w   = (const float*)d_in[13];
  const float* kan_spline_w = (const float*)d_in[14];
  const float* Wl_ue        = (const float*)d_in[18];
  const float* bl_ue        = (const float*)d_in[19];
  const float* Wr_ue        = (const float*)d_in[20];
  const float* Wl_se        = (const float*)d_in[21];
  const float* bl_se        = (const float*)d_in[22];
  const float* Wr_se        = (const float*)d_in[23];

  float* out = (float*)d_out;
  float* ws  = (float*)d_ws;
  float* he    = ws + OFF_HE;
  float* s1    = ws + OFF_S1;
  float* c1    = ws + OFF_C1;
  float* s2    = ws + OFF_S2;
  float* c2    = ws + OFF_C2;
  float* hu    = ws + OFF_HU;
  float* hs    = ws + OFF_HS;
  float* email = ws + OFF_EMAIL;
  float* bsum  = ws + OFF_BSUM;
  unsigned int* wpe16 = (unsigned int*)(ws + OFF_WPE);
  unsigned int* wl1_16 = (unsigned int*)(ws + OFF_WL1);
  unsigned int* wl2_16 = (unsigned int*)(ws + OFF_WL2);
  unsigned int* wsum16 = (unsigned int*)(ws + OFF_WS16);

  const int E1 = in_sizes[4] / 2;   // 600000 (url->email)
  const int E2 = in_sizes[5] / 2;   // 300000 (sender->email)

  k_zero<<<(int)((ZERO_N4 + 255) / 256), 256, 0, stream>>>(s1, (long)ZERO_N4);
  k_cvt_bf16<<<(KE * H / 8 + 255) / 256, 256, 0, stream>>>(Wp_email, wpe16, KE * H / 8);
  k_cvt_bf16<<<(H * H / 8 + 255) / 256, 256, 0, stream>>>(Wl_ue, wl1_16, H * H / 8);
  k_cvt_bf16<<<(H * H / 8 + 255) / 256, 256, 0, stream>>>(Wl_se, wl2_16, H * H / 8);
  k_wsum<<<(H * H / 2 + 255) / 256, 256, 0, stream>>>(Wr_ue, Wr_se, bl_ue, bl_se, wsum16, bsum);
  k_proj_url<<<(NU * H + 255) / 256, 256, 0, stream>>>(x_url, Wp_url, bp_url, hu);
  k_proj_sender<<<(NS * H + 255) / 256, 256, 0, stream>>>(x_sender, Wp_sender, bp_sender, hs);
  k_he_gemm<<<NE / 16, 256, 0, stream>>>(x_email, (const unsigned short*)wpe16, bp_email, he);
  k_scatter<<<(E1 * 32 + 255) / 256, 256, 0, stream>>>(hu, ei_rev_cont, ei_rev_cont + E1, E1, s1, c1);
  k_scatter<<<(E2 * 32 + 255) / 256, 256, 0, stream>>>(hs, ei_sends, ei_sends + E2, E2, s2, c2);
  k_combine<<<NE / 16, 256, 0, stream>>>(s1, c1, s2, c2, he,
                                         (const unsigned short*)wl1_16, (const unsigned short*)wl2_16,
                                         (const unsigned short*)wsum16, bsum, email);
  k_kan<<<(NE * 32 + 255) / 256, 256, 0, stream>>>(email, kan_base_w, kan_spline_w, out);
}
